// BigBirdBlockSparseAttention_51874615001215
// MI455X (gfx1250) — compile-verified, matrix operand order checked by reading, not run on hardware
//
#include <hip/hip_runtime.h>

// ---------------------------------------------------------------------------
// BigBird block-sparse attention for MI455X (gfx1250), bf16 WMMA pipeline.
// B=4, M=4032, H=768, NH=12, D=64, WM=64, NB=63, NW=61, R=3.
// ---------------------------------------------------------------------------

#define B_ 4
#define M_ 4032
#define H_ 768
#define NH_ 12
#define D_ 64
#define WM_ 64
#define NB_ 63
#define NW_ 61
#define R_ 3
#define NEGBIG (-10000.0f)

typedef __attribute__((ext_vector_type(16))) __bf16 v16bf;
typedef __attribute__((ext_vector_type(8)))  float  v8f;

union BFrag { v16bf v; unsigned u[8]; };

// float -> bf16 (round-to-nearest-even)
__device__ __forceinline__ unsigned short f2bf(float x) {
  unsigned u = __builtin_bit_cast(unsigned, x);
  u = (u + 0x7FFFu + ((u >> 16) & 1u)) >> 16;
  return (unsigned short)u;
}

// Load a 16x32 bf16 A/B fragment from a row-major [16+][32+] matrix.
// base points at element [tile_row0][tile_k0]; ld = row stride in elements.
// Element layout (ISA 7.12.2, 16-bit A 16x32):
//   lane L: row = L&15 ; per VGPR j: k = (j>=4?16:0) + (L>=16?8:0) + 2*(j&3)
__device__ __forceinline__ v16bf load_frag(const unsigned short* base, int ld) {
  const int lane  = threadIdx.x & 31;
  const int row   = lane & 15;
  const int khalf = (lane >> 4) << 3;   // +8 for upper half-wave
  const unsigned short* p = base + row * ld + khalf;
  BFrag f;
#pragma unroll
  for (int j = 0; j < 8; ++j) {
    const int k = ((j >= 4) ? 16 : 0) + ((j & 3) * 2);
    f.u[j] = *(const unsigned*)(p + k);   // two adjacent bf16 (k, k+1)
  }
  return f.v;
}

__device__ __forceinline__ v8f wmma_bf16(v16bf a, v16bf b, v8f c) {
  return __builtin_amdgcn_wmma_f32_16x16x32_bf16(false, a, false, b,
                                                 (short)0, c, false, false);
}

// ---------------------------------------------------------------------------
// Kernel 1: f32 -> bf16 conversion
// ---------------------------------------------------------------------------
__global__ void cvt_f32_bf16(const float* __restrict__ in,
                             unsigned short* __restrict__ out, int n) {
  int i = blockIdx.x * blockDim.x + threadIdx.x;
  if (i < n) out[i] = f2bf(in[i]);
}

// ---------------------------------------------------------------------------
// Kernel 2: QKV projection GEMM, 2-stage software pipeline (no copy rotation).
//   out[m, o] = sum_k hs[m,k] * W[o,k]      (X @ W^T)
// grid: x = M/64 (63), y = NH (o-tile == head), z = b*3 + {q,k,v}
// block: 128 threads = 4 waves; wave w owns rows [w*16, w*16+16).
// Q is pre-scaled by rsqrt(D). V additionally stored transposed as vT[d][m].
// ---------------------------------------------------------------------------
__global__ __launch_bounds__(128)
void qkv_gemm(const unsigned short* __restrict__ hs,
              const unsigned short* __restrict__ w3,
              unsigned short* __restrict__ qo,
              unsigned short* __restrict__ ko,
              unsigned short* __restrict__ vo,
              unsigned short* __restrict__ vT) {
  const int mt = blockIdx.x, h = blockIdx.y;
  const int b  = blockIdx.z / 3, wi = blockIdx.z % 3;
  const int wave = threadIdx.x >> 5, lane = threadIdx.x & 31;

  const unsigned short* W  = w3 + (size_t)wi * H_ * H_;
  const int m0 = mt * 64 + wave * 16;
  const unsigned short* A  = hs + ((size_t)b * M_ + m0) * H_;
  const unsigned short* Bp = W + (size_t)(h * 64) * H_;

  v8f acc[4];
#pragma unroll
  for (int t = 0; t < 4; ++t)
#pragma unroll
    for (int r = 0; r < 8; ++r) acc[t][r] = 0.0f;

  // Two-stage pipeline, unrolled by 2 so loads write their own register set
  // (no loop-carried register copies). H_ % 64 == 0.
  v16bf a0 = load_frag(A, H_);
  v16bf b0[4];
#pragma unroll
  for (int t = 0; t < 4; ++t) b0[t] = load_frag(Bp + (size_t)(t * 16) * H_, H_);

  for (int k0 = 0; k0 < H_; k0 += 64) {
    // stage-1 loads (k0+32 is always < H_)
    v16bf a1 = load_frag(A + k0 + 32, H_);
    v16bf b1[4];
#pragma unroll
    for (int t = 0; t < 4; ++t)
      b1[t] = load_frag(Bp + (size_t)(t * 16) * H_ + k0 + 32, H_);
    // consume stage 0
#pragma unroll
    for (int t = 0; t < 4; ++t)
      acc[t] = wmma_bf16(a0, b0[t], acc[t]);
    // stage-0 loads for next iteration (branchless in-bounds wrap on tail)
    const int k2 = (k0 + 64 < H_) ? (k0 + 64) : 0;
    a0 = load_frag(A + k2, H_);
#pragma unroll
    for (int t = 0; t < 4; ++t)
      b0[t] = load_frag(Bp + (size_t)(t * 16) * H_ + k2, H_);
    // consume stage 1
#pragma unroll
    for (int t = 0; t < 4; ++t)
      acc[t] = wmma_bf16(a1, b1[t], acc[t]);
  }

  unsigned short* dst = (wi == 0) ? qo : (wi == 1) ? ko : vo;
  const float scale = (wi == 0) ? 0.125f : 1.0f;   // rsqrt(64) folded into Q
  const int col = lane & 15, rowoff = (lane >> 4) * 8;
  const size_t bh = (size_t)b * NH_ + h;
#pragma unroll
  for (int t = 0; t < 4; ++t)
#pragma unroll
    for (int r = 0; r < 8; ++r) {
      const int m = m0 + rowoff + r;
      const int d = t * 16 + col;
      const unsigned short val = f2bf(acc[t][r] * scale);
      dst[(bh * M_ + m) * D_ + d] = val;
      if (wi == 2) vT[(bh * D_ + d) * M_ + m] = val;
    }
}

// ---------------------------------------------------------------------------
// Kernel 3: BigBird attention, flash-style over a per-q-block key-block list.
// grid: x = NB (q block), y = NH, z = B. block = 128 = 4 waves.
// Mask kinds: 0 = to_mask column bias, 1 = rand (from_blocked x to_blocked),
//             2/3/4 = band_mask window slot 0/1/2.
// ---------------------------------------------------------------------------
__global__ __launch_bounds__(128)
void bb_attn(const unsigned short* __restrict__ qb_,
             const unsigned short* __restrict__ kb_,
             const unsigned short* __restrict__ vT_,
             const float* __restrict__ band_mask,
             const float* __restrict__ from_mask,
             const float* __restrict__ to_mask,
             const float* __restrict__ from_blocked,
             const float* __restrict__ to_blocked,
             const int*   __restrict__ rand_attn,
             float* __restrict__ out) {
  const int qb = blockIdx.x, h = blockIdx.y, b = blockIdx.z;
  const int wave = threadIdx.x >> 5, lane = threadIdx.x & 31;
  const int col = lane & 15, rowoff = (lane >> 4) * 8;

  __shared__ int s_blk[NB_];
  __shared__ int s_kind[NB_];
  __shared__ int s_n;
  __shared__ unsigned short s_p[4][16 * 64];   // per-wave P staging (bf16)

  if (threadIdx.x == 0) {
    int n = 0;
    if (qb == 0 || qb == NB_ - 1) {
      for (int j = 0; j < NB_; ++j) { s_blk[n] = j; s_kind[n] = 0; ++n; }
    } else {
      const int* ra = rand_attn + (((size_t)b * NH_ + h) * NW_ + (qb - 1)) * R_;
      if (qb == 1) {
        const int fix[4] = {0, 1, 2, NB_ - 1};
        for (int j = 0; j < 4; ++j) { s_blk[n] = fix[j]; s_kind[n] = 0; ++n; }
      } else if (qb == NB_ - 2) {
        const int fix[4] = {0, NB_ - 3, NB_ - 2, NB_ - 1};
        for (int j = 0; j < 4; ++j) { s_blk[n] = fix[j]; s_kind[n] = 0; ++n; }
      } else {
        s_blk[n] = 0;      s_kind[n] = 0; ++n;           // global first block
        s_blk[n] = qb - 1; s_kind[n] = 2; ++n;           // band window 0
        s_blk[n] = qb;     s_kind[n] = 3; ++n;           // band window 1
        s_blk[n] = qb + 1; s_kind[n] = 4; ++n;           // band window 2
        s_blk[n] = NB_ - 1; s_kind[n] = 0; ++n;          // global last block
      }
      for (int j = 0; j < R_; ++j) { s_blk[n] = ra[j]; s_kind[n] = 1; ++n; }
    }
    s_n = n;
  }
  __syncthreads();

  const size_t bh = (size_t)b * NH_ + h;
  const unsigned short* Qp = qb_ + (bh * M_ + qb * 64 + wave * 16) * D_;
  const v16bf aQ0 = load_frag(Qp, D_);
  const v16bf aQ1 = load_frag(Qp + 32, D_);

  float mrun[8], lrun[8], fbrow[8];
  v8f acc[4];
#pragma unroll
  for (int r = 0; r < 8; ++r) {
    mrun[r] = -1e30f; lrun[r] = 0.0f;
    fbrow[r] = from_blocked[((size_t)b * NB_ + qb) * WM_ + wave * 16 + rowoff + r];
  }
#pragma unroll
  for (int t = 0; t < 4; ++t)
#pragma unroll
    for (int r = 0; r < 8; ++r) acc[t][r] = 0.0f;

  const int nlist = s_n;
  for (int i = 0; i < nlist; ++i) {
    const int kblk = s_blk[i];
    const int kind = s_kind[i];

    // prefetch next key block into cache (gfx1250 global_prefetch_b8)
    if (i + 1 < nlist) {
      const unsigned short* Kn = kb_ + (bh * M_ + s_blk[i + 1] * 64) * D_;
      __builtin_prefetch(Kn + (lane << 6), 0, 1);          // rows 0..31
      __builtin_prefetch(Kn + ((lane + 32) << 6), 0, 1);   // rows 32..63
    }

    // ---- load all K fragments up-front (batched b128 clause) ----
    const unsigned short* Kp = kb_ + (bh * M_ + kblk * 64) * D_;
    v16bf bK[8];
#pragma unroll
    for (int t = 0; t < 4; ++t) {
      bK[2 * t]     = load_frag(Kp + (size_t)(t * 16) * D_, D_);
      bK[2 * t + 1] = load_frag(Kp + (size_t)(t * 16) * D_ + 32, D_);
    }

    // ---- scores S = Q * K^T (16 rows x 64 keys), 8 WMMAs ----
    v8f s[4];
#pragma unroll
    for (int t = 0; t < 4; ++t) {
      v8f c;
#pragma unroll
      for (int r = 0; r < 8; ++r) c[r] = 0.0f;
      c = wmma_bf16(aQ0, bK[2 * t], c);
      c = wmma_bf16(aQ1, bK[2 * t + 1], c);
      s[t] = c;
    }

    // ---- issue V fragment loads now; latency hidden under softmax VALU ----
    const unsigned short* Vp = vT_ + (bh * D_) * (size_t)M_ + kblk * 64;
    v16bf bV[8];
#pragma unroll
    for (int t = 0; t < 4; ++t) {
      bV[2 * t]     = load_frag(Vp + (size_t)(t * 16) * M_, M_);
      bV[2 * t + 1] = load_frag(Vp + (size_t)(t * 16) * M_ + 32, M_);
    }

    // ---- mask bias ----
#pragma unroll
    for (int t = 0; t < 4; ++t) {
      const int kin = t * 16 + col;            // key within block
      const int kglob = kblk * 64 + kin;       // global key index
      if (kind == 0) {
        const float bias = (1.0f - to_mask[(size_t)b * M_ + kglob]) * NEGBIG;
#pragma unroll
        for (int r = 0; r < 8; ++r) s[t][r] += bias;
      } else if (kind == 1) {
        const float tb = to_blocked[((size_t)b * NB_ + kblk) * WM_ + kin];
#pragma unroll
        for (int r = 0; r < 8; ++r)
          s[t][r] += (1.0f - fbrow[r] * tb) * NEGBIG;
      } else {
        const int wi = kind - 2;
        const float* bm = band_mask +
            (((size_t)b * (NB_ - 4) + (qb - 2)) * WM_) * (3 * WM_);
#pragma unroll
        for (int r = 0; r < 8; ++r) {
          const int qr = wave * 16 + rowoff + r;
          s[t][r] += (1.0f - bm[(size_t)qr * (3 * WM_) + wi * 64 + kin]) * NEGBIG;
        }
      }
    }

    // ---- online softmax: row max across 16 lanes of each half ----
    float mnew[8];
#pragma unroll
    for (int r = 0; r < 8; ++r) {
      float v = fmaxf(fmaxf(s[0][r], s[1][r]), fmaxf(s[2][r], s[3][r]));
#pragma unroll
      for (int off = 8; off >= 1; off >>= 1)
        v = fmaxf(v, __shfl_xor(v, off, 32));
      mnew[r] = fmaxf(mrun[r], v);
    }
#pragma unroll
    for (int r = 0; r < 8; ++r) {
      const float corr = __expf(mrun[r] - mnew[r]);
      lrun[r] *= corr;
#pragma unroll
      for (int t = 0; t < 4; ++t) acc[t][r] *= corr;
      mrun[r] = mnew[r];
    }
#pragma unroll
    for (int t = 0; t < 4; ++t)
#pragma unroll
      for (int r = 0; r < 8; ++r)
        s[t][r] = __expf(s[t][r] - mnew[r]);
#pragma unroll
    for (int r = 0; r < 8; ++r) {
      float v = s[0][r] + s[1][r] + s[2][r] + s[3][r];
#pragma unroll
      for (int off = 8; off >= 1; off >>= 1)
        v += __shfl_xor(v, off, 32);
      lrun[r] += v;
    }

    // ---- stage P (C-layout -> A-fragment layout) through LDS ----
    unsigned short* P = &s_p[wave][0];
#pragma unroll
    for (int t = 0; t < 4; ++t)
#pragma unroll
      for (int r = 0; r < 8; ++r)
        P[(rowoff + r) * 64 + t * 16 + col] = f2bf(s[t][r]);
    // LDS ops from the same wave are in-order; region is wave-private.
    const v16bf aP0 = load_frag(P, 64);
    const v16bf aP1 = load_frag(P + 32, 64);

    // ---- context += P * V  (V^T gives contiguous keys), 8 WMMAs ----
#pragma unroll
    for (int t = 0; t < 4; ++t) {
      acc[t] = wmma_bf16(aP0, bV[2 * t], acc[t]);
      acc[t] = wmma_bf16(aP1, bV[2 * t + 1], acc[t]);
    }
  }

  // ---- epilogue: normalize, apply from_mask, store f32 output ----
#pragma unroll
  for (int t = 0; t < 4; ++t)
#pragma unroll
    for (int r = 0; r < 8; ++r) {
      const int m = qb * 64 + wave * 16 + rowoff + r;
      const float o = acc[t][r] / lrun[r] * from_mask[(size_t)b * M_ + m];
      out[((size_t)b * M_ + m) * H_ + h * 64 + t * 16 + col] = o;
    }
}

// ---------------------------------------------------------------------------
// kernel_launch
// inputs: hidden_states, Wq, Wk, Wv, band_mask, from_mask, to_mask,
//         from_blocked_mask, to_blocked_mask, rand_attn
// ---------------------------------------------------------------------------
extern "C" void kernel_launch(void* const* d_in, const int* in_sizes, int n_in,
                              void* d_out, int out_size, void* d_ws, size_t ws_size,
                              hipStream_t stream) {
  const float* hs = (const float*)d_in[0];
  const float* Wq = (const float*)d_in[1];
  const float* Wk = (const float*)d_in[2];
  const float* Wv = (const float*)d_in[3];
  const float* band_mask    = (const float*)d_in[4];
  const float* from_mask    = (const float*)d_in[5];
  const float* to_mask      = (const float*)d_in[6];
  const float* from_blocked = (const float*)d_in[7];
  const float* to_blocked   = (const float*)d_in[8];
  const int*   rand_attn    = (const int*)d_in[9];
  float* out = (float*)d_out;

  const size_t nHS = (size_t)B_ * M_ * H_;   // 12,386,304
  const size_t nW  = (size_t)H_ * H_;        // 589,824

  unsigned short* hsb = (unsigned short*)d_ws;
  unsigned short* wb  = hsb + nHS;           // 3 weights, bf16
  unsigned short* qb  = wb + 3 * nW;
  unsigned short* kb  = qb + nHS;
  unsigned short* vb  = kb + nHS;
  unsigned short* vT  = vb + nHS;

  const int thr = 256;
  cvt_f32_bf16<<<(int)((nHS + thr - 1) / thr), thr, 0, stream>>>(hs, hsb, (int)nHS);
  cvt_f32_bf16<<<(int)((nW + thr - 1) / thr), thr, 0, stream>>>(Wq, wb, (int)nW);
  cvt_f32_bf16<<<(int)((nW + thr - 1) / thr), thr, 0, stream>>>(Wk, wb + nW, (int)nW);
  cvt_f32_bf16<<<(int)((nW + thr - 1) / thr), thr, 0, stream>>>(Wv, wb + 2 * nW, (int)nW);

  qkv_gemm<<<dim3(M_ / 64, NH_, B_ * 3), 128, 0, stream>>>(hsb, wb, qb, kb, vb, vT);

  bb_attn<<<dim3(NB_, NH_, B_), 128, 0, stream>>>(
      qb, kb, vT, band_mask, from_mask, to_mask,
      from_blocked, to_blocked, rand_attn, out);
}
